// MSDeformAttnKDA_81295140979208
// MI455X (gfx1250) — matrix operand order repaired and checked by reading, not compiled, not run on hardware
//
#include <hip/hip_runtime.h>
#include <hip/hip_bf16.h>
#include <math.h>

typedef __attribute__((ext_vector_type(16))) __bf16 v16bf;
typedef __attribute__((ext_vector_type(8)))  __bf16 v8bf;
typedef __attribute__((ext_vector_type(8)))  float  v8f;
typedef __attribute__((ext_vector_type(4)))  unsigned int u32x4;
typedef __attribute__((ext_vector_type(8)))  int    i32x8;
typedef __attribute__((ext_vector_type(4)))  int    i32x4;

#define BS    2
#define LQ    13125
#define CC    256
#define MHEAD 8
#define DHEAD 32
#define NLVL  3
#define NPTS  4
#define NOFF  (MHEAD*NLVL*NPTS*2)   /* 192 */
#define ROWS  (BS*LQ)               /* 26250 */
#define KB    (CC/32)               /* 8 k-blocks per GEMM */

// ---------------------------------------------------------------------------
// Stage A: fp32 -> bf16 conversion (elementwise, activations only)
// ---------------------------------------------------------------------------
__global__ void cvt_f32_bf16(const float* __restrict__ in, __bf16* __restrict__ out, int n) {
  int i = blockIdx.x * blockDim.x + threadIdx.x;
  if (i < n) out[i] = (__bf16)in[i];
}

// ---------------------------------------------------------------------------
// Pre-pack W[256,N] f32 into per-wave WMMA B fragments (bf16):
//   Wp[((nt*KB + kb)*32 + lane)*16 + e] = W[kB*N + n]
//   kB = kb*32 + (lane>>4)*16 + e  (B 32x16 layout: half-waves split K)
//   n  = nt*16 + (lane&15)
// => in the GEMM each lane loads its 16 bf16 fragment contiguously (2xb128).
// ---------------------------------------------------------------------------
__global__ void pack_w_bf16(const float* __restrict__ W, __bf16* __restrict__ Wp, int N) {
  int i = blockIdx.x * blockDim.x + threadIdx.x;
  int total = (N / 16) * KB * 32 * 16;
  if (i >= total) return;
  int e    = i & 15;
  int lane = (i >> 4) & 31;
  int rest = i >> 9;
  int kb   = rest & (KB - 1);
  int nt   = rest >> 3;              // KB == 8
  int kB   = kb * 32 + (lane >> 4) * 16 + e;
  int n    = nt * 16 + (lane & 15);
  Wp[i] = (__bf16)W[(size_t)kB * N + n];
}

// ---------------------------------------------------------------------------
// Stage B/D: C[rows,N] = A[rows,256](bf16) @ W(packed bf16) + bias[N]
// A tile (16x256 bf16 = 8KB, memory-contiguous) staged to LDS via the
// Tensor Data Mover; 8 x v_wmma_f32_16x16x32_bf16 per tile.
// ---------------------------------------------------------------------------
__global__ void gemm_bf16_wmma(const __bf16* __restrict__ A,
                               const __bf16* __restrict__ Wp,
                               const float*  __restrict__ bias,
                               float* __restrict__ C,
                               int rows, int N) {
  __shared__ __bf16 lA[16 * CC];                       // 8 KB
  const int tileM = blockIdx.x * 16;
  const int wave  = threadIdx.x >> 5;
  const int lane  = threadIdx.x & 31;

#if __has_builtin(__builtin_amdgcn_tensor_load_to_lds) && __has_builtin(__builtin_amdgcn_s_wait_tensorcnt)
  if (wave == 0) {
    // Elements remaining from tile start; TDM OOB reads return zero for the
    // row tail (and any misread tail rows are never stored anyway).
    unsigned long long remLL = (unsigned long long)rows * CC - (unsigned long long)tileM * CC;
    unsigned rem = (remLL > 16u * CC) ? 16u * CC : (unsigned)remLL;
    unsigned long long ga = (unsigned long long)(uintptr_t)(A + (size_t)tileM * CC);
    unsigned lds_off = (unsigned)(uintptr_t)(&lA[0]);

    u32x4 g0;
    g0[0] = 1u;                                        // count=1, user descriptor
    g0[1] = lds_off;                                   // lds_addr
    g0[2] = (unsigned)(ga & 0xFFFFFFFFu);              // global_addr[31:0]
    g0[3] = (unsigned)((ga >> 32) & 0x1FFFFFFu) | (2u << 30);  // addr[56:32] | type=2

    i32x8 g1;
    g1[0] = 0x00010000;                                // data_size=1 (2 bytes/elem)
    g1[1] = (int)((rem & 0xFFFFu) << 16);              // tensor_dim0[15:0]
    g1[2] = (int)((rem >> 16) | (1u << 16));           // tensor_dim0[31:16] | tensor_dim1=1
    g1[3] = (int)(4096u << 16);                        // tile_dim0 = 4096 elements
    g1[4] = 1;                                         // tile_dim1 = 1
    g1[5] = (int)rem;                                  // tensor_dim0_stride[31:0]
    g1[6] = 0;
    g1[7] = 0;
    i32x4 z4 = {0, 0, 0, 0};
    i32x8 z8 = {0, 0, 0, 0, 0, 0, 0, 0};
    __builtin_amdgcn_tensor_load_to_lds(g0, g1, z4, z4, z8, 0);
    __builtin_amdgcn_s_wait_tensorcnt(0);
  }
#else
  for (int idx = threadIdx.x; idx < 16 * CC; idx += blockDim.x) {
    int gm = tileM + (idx >> 8);
    lA[idx] = (gm < rows) ? A[(size_t)gm * CC + (idx & 255)] : (__bf16)0.0f;
  }
#endif
  __syncthreads();

  const int n0 = (blockIdx.y * 4 + wave) * 16;
  if (n0 >= N) return;                                 // wave-uniform exit
  const int hl  = lane >> 4;
  const int l15 = lane & 15;

  // packed B fragment pointer for this wave/lane; advances 32*16 elems per kb
  const __bf16* wp = Wp + ((size_t)(n0 >> 4) * KB * 32 + lane) * 16;

  v8f acc = {};
  #pragma unroll
  for (int kb = 0; kb < KB; ++kb) {
    // A 16x32 bf16 fragment: one contiguous 16-element run per lane (k = hl*16 + e)
    const __bf16* arow = &lA[l15 * CC + kb * 32 + hl * 16];
    v8bf alo = *(const v8bf*)(arow);
    v8bf ahi = *(const v8bf*)(arow + 8);
    v16bf amat = __builtin_shufflevector(alo, ahi,
        0, 1, 2, 3, 4, 5, 6, 7, 8, 9, 10, 11, 12, 13, 14, 15);
    v16bf bmat = *(const v16bf*)(wp + (size_t)kb * 32 * 16);
    acc = __builtin_amdgcn_wmma_f32_16x16x32_bf16(
        /*neg_a=*/false, amat, /*neg_b=*/false, bmat,
        /*c_mod=*/(short)0, acc, /*reuse_a=*/false, /*reuse_b=*/false);
  }

  // D 16x16 f32: VGPR r -> M = hl*8 + r, lane -> N = n0 + l15
  #pragma unroll
  for (int r = 0; r < 8; ++r) {
    int m = tileM + hl * 8 + r;
    if (m < rows) C[(size_t)m * N + n0 + l15] = acc[r] + bias[n0 + l15];
  }
}

// ---------------------------------------------------------------------------
// Stage C: bilinear sampling of key/value at L*P=12 points, q.k logits with
// cross-lane reduction, softmax over 12, weighted value sum.
// One wave32 per (b,q,m); lane = channel d in [0,32). Corner gathers are
// coalesced across the wave (32 consecutive floats).
// ---------------------------------------------------------------------------
__global__ void msda_sample_attn(const float* __restrict__ value,
                                 const float* __restrict__ keyt,
                                 const float* __restrict__ qp,
                                 const float* __restrict__ offs,
                                 const float* __restrict__ refb,
                                 float* __restrict__ omid) {
  const int gw   = blockIdx.x * (blockDim.x >> 5) + (threadIdx.x >> 5);
  const int lane = threadIdx.x & 31;
  if (gw >= BS * LQ * MHEAD) return;

  const int m = gw & 7;
  const int t = gw >> 3;
  const int q = t % LQ;
  const int b = t / LQ;

  const int lvlH[NLVL]    = {100, 50, 25};
  const int lvlW[NLVL]    = {100, 50, 25};
  const int lvlBase[NLVL] = {0, 10000, 12500};

  const float* vbase = value + (size_t)b * LQ * CC + m * DHEAD + lane;
  const float* kbase = keyt  + (size_t)b * LQ * CC + m * DHEAD + lane;
  const float  qd = qp[((size_t)b * LQ + q) * CC + m * DHEAD + lane]
                    * 0.17677669529663687f;            // 1/sqrt(D=32)
  const float* rb = refb + ((size_t)b * LQ + q) * (NLVL * 2);
  const float* of = offs + ((size_t)b * LQ + q) * NOFF + m * (NLVL * NPTS * 2);

  float logit[NLVL * NPTS];
  float vsamp[NLVL * NPTS];

  #pragma unroll
  for (int l = 0; l < NLVL; ++l) {
    const int   Wi = lvlW[l], Hi = lvlH[l], base = lvlBase[l];
    const float Wf = (float)Wi, Hf = (float)Hi;
    const float rx = rb[l * 2 + 0], ry = rb[l * 2 + 1];
    #pragma unroll
    for (int p = 0; p < NPTS; ++p) {
      const int pi = l * NPTS + p;
      float gx = rx * Wf + of[pi * 2 + 0] - 0.5f;      // loc*W - 0.5
      float gy = ry * Hf + of[pi * 2 + 1] - 0.5f;
      float x0f = floorf(gx), y0f = floorf(gy);
      int   x0 = (int)x0f, y0 = (int)y0f;
      int   x1 = x0 + 1,   y1 = y0 + 1;
      float wx1 = gx - x0f, wy1 = gy - y0f;
      float wx0 = 1.0f - wx1, wy0 = 1.0f - wy1;

      float kacc = 0.0f, vacc = 0.0f;
      const int   cx[4] = {x0, x1, x0, x1};
      const int   cy[4] = {y0, y0, y1, y1};
      const float cw[4] = {wx0 * wy0, wx1 * wy0, wx0 * wy1, wx1 * wy1};
      #pragma unroll
      for (int c = 0; c < 4; ++c) {
        int xi = cx[c], yi = cy[c];
        if (xi >= 0 && xi < Wi && yi >= 0 && yi < Hi) {
          size_t pos = (size_t)(base + yi * Wi + xi) * CC;
          kacc += cw[c] * kbase[pos];
          vacc += cw[c] * vbase[pos];
        }
      }
      float part = qd * kacc;
      #pragma unroll
      for (int s = 16; s >= 1; s >>= 1) part += __shfl_xor(part, s, 32);
      logit[pi] = part;
      vsamp[pi] = vacc;
    }
  }

  float mx = logit[0];
  #pragma unroll
  for (int i = 1; i < NLVL * NPTS; ++i) mx = fmaxf(mx, logit[i]);
  float ex[NLVL * NPTS];
  float se = 0.0f;
  #pragma unroll
  for (int i = 0; i < NLVL * NPTS; ++i) { ex[i] = __expf(logit[i] - mx); se += ex[i]; }
  const float inv = 1.0f / se;

  float o = 0.0f;
  #pragma unroll
  for (int i = 0; i < NLVL * NPTS; ++i) o += ex[i] * inv * vsamp[i];

  omid[((size_t)b * LQ + q) * CC + m * DHEAD + lane] = o;
}

// ---------------------------------------------------------------------------
// Host-side orchestration
// ---------------------------------------------------------------------------
static inline size_t alignup(size_t x) { return (x + 255) & ~(size_t)255; }

extern "C" void kernel_launch(void* const* d_in, const int* in_sizes, int n_in,
                              void* d_out, int out_size, void* d_ws, size_t ws_size,
                              hipStream_t stream) {
  const float* query = (const float*)d_in[0];
  const float* refb  = (const float*)d_in[1];
  const float* inpf  = (const float*)d_in[2];
  const float* Wv    = (const float*)d_in[3];
  const float* bv    = (const float*)d_in[4];
  const float* Wk    = (const float*)d_in[5];
  const float* bk    = (const float*)d_in[6];
  const float* Wq    = (const float*)d_in[7];
  const float* bq    = (const float*)d_in[8];
  const float* Woff  = (const float*)d_in[9];
  const float* boff  = (const float*)d_in[10];
  const float* Wout  = (const float*)d_in[11];
  const float* bout  = (const float*)d_in[12];
  float* out = (float*)d_out;

  char* ws = (char*)d_ws;
  size_t off = 0;
  auto carve = [&](size_t bytes) { char* p = ws + off; off += alignup(bytes); return p; };

  __bf16* bf_in   = (__bf16*)carve((size_t)ROWS * CC * 2);
  __bf16* bf_q    = (__bf16*)carve((size_t)ROWS * CC * 2);
  __bf16* wv_p    = (__bf16*)carve((size_t)16 * KB * 32 * 16 * 2);   // N=256 packs
  __bf16* wk_p    = (__bf16*)carve((size_t)16 * KB * 32 * 16 * 2);
  __bf16* wq_p    = (__bf16*)carve((size_t)16 * KB * 32 * 16 * 2);
  __bf16* wout_p  = (__bf16*)carve((size_t)16 * KB * 32 * 16 * 2);
  __bf16* woff_p  = (__bf16*)carve((size_t)12 * KB * 32 * 16 * 2);   // N=192 pack
  float*  val     = (float*) carve((size_t)ROWS * CC * 4);
  float*  keyp    = (float*) carve((size_t)ROWS * CC * 4);
  float*  qproj   = (float*) carve((size_t)ROWS * CC * 4);
  float*  offv    = (float*) carve((size_t)ROWS * NOFF * 4);
  float*  omid    = (float*) carve((size_t)ROWS * CC * 4);
  __bf16* bf_omid = (__bf16*)carve((size_t)ROWS * CC * 2);
  (void)ws_size; (void)in_sizes; (void)n_in; (void)out_size;

  const int nAct = ROWS * CC;                          // 6.72M
  auto cvt = [&](const float* src, __bf16* dst, int n) {
    cvt_f32_bf16<<<(n + 255) / 256, 256, 0, stream>>>(src, dst, n);
  };
  auto pack = [&](const float* W, __bf16* Wp, int N) {
    int total = (N / 16) * KB * 32 * 16;
    pack_w_bf16<<<(total + 255) / 256, 256, 0, stream>>>(W, Wp, N);
  };

  // Stage A: conversions + weight fragment packing
  cvt(inpf,  bf_in, nAct);
  cvt(query, bf_q,  nAct);
  pack(Wv,   wv_p,   CC);
  pack(Wk,   wk_p,   CC);
  pack(Wq,   wq_p,   CC);
  pack(Wout, wout_p, CC);
  pack(Woff, woff_p, NOFF);

  // Stage B: projection GEMMs (wave32, 4 waves/block, 16x16 tile/wave)
  const int rowTiles = (ROWS + 15) / 16;               // 1641
  dim3 blk(128);
  dim3 g256(rowTiles, CC / 64);                        // N=256 -> y=4
  dim3 g192(rowTiles, NOFF / 64);                      // N=192 -> y=3
  gemm_bf16_wmma<<<g256, blk, 0, stream>>>(bf_in, wv_p,   bv,   val,   ROWS, CC);
  gemm_bf16_wmma<<<g256, blk, 0, stream>>>(bf_in, wk_p,   bk,   keyp,  ROWS, CC);
  gemm_bf16_wmma<<<g256, blk, 0, stream>>>(bf_q,  wq_p,   bq,   qproj, ROWS, CC);
  gemm_bf16_wmma<<<g192, blk, 0, stream>>>(bf_q,  woff_p, boff, offv,  ROWS, NOFF);

  // Stage C: sampling + attention (1 wave per (b,q,m); 4 waves per block)
  const int items = BS * LQ * MHEAD;                   // 210000 waves
  msda_sample_attn<<<(items + 3) / 4, 128, 0, stream>>>(val, keyp, qproj, offv, refb, omid);

  // Stage D: output projection
  cvt(omid, bf_omid, nAct);
  gemm_bf16_wmma<<<g256, blk, 0, stream>>>(bf_omid, wout_p, bout, out, ROWS, CC);
}
